// MultiHeadAttention_47321949667756
// MI455X (gfx1250) — matrix operand order repaired and checked by reading, not compiled, not run on hardware
//
#include <hip/hip_runtime.h>
#include <hip/hip_bf16.h>

// ---------------------------------------------------------------------------
// MHA forward for MI455X (gfx1250, wave32).
//  - All matmuls on v_wmma_f32_16x16x32_bf16 (fp32 accumulate).
//  - Flash attention: double-buffered K/V staging via GLOBAL_LOAD_ASYNC_TO_LDS
//    (ASYNCcnt overlaps DMA of block kb+1 with WMMA compute of block kb);
//    each wave processes 32 queries against the shared staged block.
//  - GEMM waves compute 16x128 tiles (8 accumulators).
// ---------------------------------------------------------------------------

typedef __bf16 bf16;
typedef __attribute__((ext_vector_type(16))) __bf16 v16bf;
typedef __attribute__((ext_vector_type(8)))  float  v8f;

constexpr int Bb = 2;
constexpr int Tt = 4096;
constexpr int Ee = 1024;
constexpr int Hh = 16;
constexpr int Dd = 64;
constexpr int N3 = 3 * Ee;           // 3072
constexpr int Mrows = Bb * Tt;       // 8192

#define WMMA_BF16(a, b, c)                                                    \
  __builtin_amdgcn_wmma_f32_16x16x32_bf16(false, (a), false, (b), (short)0,   \
                                          (c), false, false)

// K-index inside a 16-bit 16x32 A operand: element e (0..15), lane-half h.
// ISA 7.12.2: lanes 0-15 hold K {0..7,16..23}, lanes 16-31 hold K {8..15,24..31}.
__device__ __forceinline__ int a_k(int e, int h) {
  return (e < 8 ? e : e + 8) + 8 * h;
}

__device__ __forceinline__ v8f zero8() {
  v8f z;
#pragma unroll
  for (int i = 0; i < 8; ++i) z[i] = 0.0f;
  return z;
}

// Async global->LDS copy, 16 bytes per lane (ASYNCcnt-tracked, no VGPRs).
__device__ __forceinline__ void async_ld_b128(unsigned lds_off,
                                              const void* gaddr) {
  asm volatile("global_load_async_to_lds_b128 %0, %1, off" ::"v"(lds_off),
               "v"(gaddr)
               : "memory");
}

__device__ __forceinline__ unsigned lds_off_of(const void* p) {
  // Generic LDS pointer: addr[31:0] is the LDS byte offset (ISA 10.2).
  return (unsigned)(unsigned long long)p;
}

// ---------------------------------------------------------------------------
// Kernel 0: fp32 -> bf16 weight conversion (one-shot, bandwidth-trivial).
// ---------------------------------------------------------------------------
__global__ void cvt_weights(const float* __restrict__ wq,
                            const float* __restrict__ wo,
                            bf16* __restrict__ wqb, bf16* __restrict__ wob) {
  const int n1 = Ee * N3;
  const int tot = n1 + Ee * Ee;
  for (int i = blockIdx.x * blockDim.x + threadIdx.x; i < tot;
       i += gridDim.x * blockDim.x) {
    if (i < n1)
      wqb[i] = (bf16)wq[i];
    else
      wob[i - n1] = (bf16)wo[i - n1];
  }
}

// ---------------------------------------------------------------------------
// Kernel 1: qkv = x @ w_qkv + b_qkv, scattered into Q/K/V (B,H,T,D) bf16.
// One wave computes a 16(M) x 128(N) tile; K loop in 32-wide chunks.
// ---------------------------------------------------------------------------
__global__ __launch_bounds__(256) void qkv_gemm(
    const float* __restrict__ x, const bf16* __restrict__ wqb,
    const float* __restrict__ bias, bf16* __restrict__ Q,
    bf16* __restrict__ K, bf16* __restrict__ V) {
  const int lane = threadIdx.x & 31;
  const int wave = threadIdx.x >> 5;
  const int W = blockIdx.x * 8 + wave;
  const int NT = N3 / 128;  // 24 n-groups
  const int mt = W / NT, ng = W % NT;
  const int m0 = mt * 16, n0 = ng * 128;
  const int h = lane >> 4, ml = lane & 15;
  const int row = m0 + ml;

  v8f acc[8];
#pragma unroll
  for (int t = 0; t < 8; ++t) acc[t] = zero8();

  for (int kc = 0; kc < Ee / 32; ++kc) {
    v16bf a;
#pragma unroll
    for (int e = 0; e < 16; ++e)
      a[e] = (bf16)x[(size_t)row * Ee + kc * 32 + a_k(e, h)];

    const bf16* bp = wqb + (size_t)(kc * 32) * N3 + n0 + ml;
#pragma unroll
    for (int t = 0; t < 8; ++t) {
      v16bf b;
#pragma unroll
      for (int e = 0; e < 16; ++e) b[e] = bp[(size_t)a_k(e, h) * N3 + t * 16];
      acc[t] = WMMA_BF16(a, b, acc[t]);
    }
  }

#pragma unroll
  for (int t = 0; t < 8; ++t) {
    const int n = n0 + t * 16 + ml;          // column handled by this lane
    const float bv = bias[n];
    const int hh = n / (3 * Dd);
    const int r3 = n % (3 * Dd);
    const int which = r3 / Dd;
    const int d = r3 % Dd;
    bf16* dst = which == 0 ? Q : which == 1 ? K : V;
#pragma unroll
    for (int r = 0; r < 8; ++r) {
      const int gm = m0 + r + 8 * h;         // global row (b*T + t)
      const int bidx = gm / Tt, tt = gm % Tt;
      const size_t o = (((size_t)bidx * Hh + hh) * Tt + tt) * Dd + d;
      dst[o] = (bf16)(acc[t][r] + bv);
    }
  }
}

// ---------------------------------------------------------------------------
// Kernel 2: causal flash attention.
// Block = 4 waves sharing (b,h); each wave owns 32 queries (two 16-row A
// tiles). Key loop in 32-key blocks with double-buffered async K/V staging:
//   prefetch(kb+1) -> LDS buffer ^1 overlaps WMMA compute on buffer kb&1.
// Per block iter & wave: 8 WMMAs for S, fp32 online softmax, P re-swizzled
// via LDS into A layout, 8 WMMAs for O += P*V.
// ---------------------------------------------------------------------------
__global__ __launch_bounds__(128) void flash_attn(
    const bf16* __restrict__ Q, const bf16* __restrict__ K,
    const bf16* __restrict__ V, bf16* __restrict__ O) {
  __shared__ bf16 Ks[2][32 * 64];    // 2 x 4 KB: 32 keys x 64 d, ping-pong
  __shared__ bf16 Vs[2][32 * 64];    // 2 x 4 KB
  __shared__ bf16 pbuf[4][16 * 32];  // 4 x 1 KB: per-wave P tile (reused)
  const int tid = threadIdx.x;
  const int lane = tid & 31;
  const int wave = tid >> 5;
  const int bh = blockIdx.y;                  // b*H + head
  const int q0 = (blockIdx.x * 4 + wave) * 32;
  const int h = lane >> 4, ml = lane & 15;
  bf16* pl = pbuf[wave];

  const bf16* Qb = Q + (size_t)bh * Tt * Dd;
  const bf16* Kb = K + (size_t)bh * Tt * Dd;
  const bf16* Vb = V + (size_t)bh * Tt * Dd;

  // Q as A operands: 2 query sub-tiles x 2 d-chunks.
  v16bf qa[2][2];
#pragma unroll
  for (int qt = 0; qt < 2; ++qt) {
    const bf16* qp = Qb + (size_t)(q0 + qt * 16 + ml) * Dd;
#pragma unroll
    for (int e = 0; e < 16; ++e) {
      qa[qt][0][e] = qp[a_k(e, h)];
      qa[qt][1][e] = qp[32 + a_k(e, h)];
    }
  }

  float Mr[2][8], Lr[2][8];
  v8f oacc[2][4];
#pragma unroll
  for (int qt = 0; qt < 2; ++qt) {
#pragma unroll
    for (int r = 0; r < 8; ++r) {
      Mr[qt][r] = -1e30f;
      Lr[qt][r] = 0.0f;
    }
#pragma unroll
    for (int nt = 0; nt < 4; ++nt) oacc[qt][nt] = zero8();
  }

  const int q0max = (blockIdx.x * 4 + 3) * 32;
  const int nblk = (q0max + 32 + 31) / 32;  // block-wide causal range

  // Stage key-block kb into LDS buffer `buf` (contiguous 4 KB each).
  auto stage = [&](int kb, int buf) {
    const char* kg = (const char*)(Kb + (size_t)kb * 32 * Dd);
    const char* vg = (const char*)(Vb + (size_t)kb * 32 * Dd);
    const unsigned kso = lds_off_of(&Ks[buf][0]);
    const unsigned vso = lds_off_of(&Vs[buf][0]);
#pragma unroll
    for (int j = 0; j < 2; ++j) {
      const unsigned off = (unsigned)(tid + 128 * j) * 16;  // 16B chunks
      async_ld_b128(kso + off, kg + off);
      async_ld_b128(vso + off, vg + off);
    }
  };

  stage(0, 0);
  for (int kb = 0; kb < nblk; ++kb) {
    const int key0 = kb * 32;
    const int cur = kb & 1;

    asm volatile("s_wait_asynccnt 0" ::: "memory");  // my stage(kb) landed
    __syncthreads();  // everyone's stage(kb) landed; buf[cur^1] free to fill
    if (kb + 1 < nblk) stage(kb + 1, cur ^ 1);       // overlaps compute below

    const bf16* ks = &Ks[cur][0];
    const bf16* vs = &Vs[cur][0];

#pragma unroll
    for (int qt = 0; qt < 2; ++qt) {
      const int qbase = q0 + qt * 16;
      if (key0 > qbase + 15) continue;  // tile fully masked (uniform per wave)

      // ---- S = Q * K^T, two key sub-tiles; B operand from LDS ----
      v8f s0 = zero8(), s1 = zero8();
#pragma unroll
      for (int c = 0; c < 2; ++c) {
        v16bf kt0, kt1;
#pragma unroll
        for (int e = 0; e < 16; ++e) {
          kt0[e] = ks[ml * 64 + c * 32 + a_k(e, h)];
          kt1[e] = ks[(16 + ml) * 64 + c * 32 + a_k(e, h)];
        }
        s0 = WMMA_BF16(qa[qt][c], kt0, s0);
        s1 = WMMA_BF16(qa[qt][c], kt1, s1);
      }

      // ---- scale + causal mask (C/D: row = r+8h, col = ml) ----
#pragma unroll
      for (int r = 0; r < 8; ++r) {
        const int qrow = qbase + r + 8 * h;
        float v0 = s0[r] * 0.125f;  // 1/sqrt(64)
        float v1 = s1[r] * 0.125f;
        if (key0 + ml > qrow) v0 = -1e30f;
        if (key0 + 16 + ml > qrow) v1 = -1e30f;
        s0[r] = v0;
        s1[r] = v1;
      }

      // ---- online softmax: row stats via 16-lane xor reductions ----
      float alpha[8];
#pragma unroll
      for (int r = 0; r < 8; ++r) {
        float mx = fmaxf(s0[r], s1[r]);
#pragma unroll
        for (int msk = 1; msk < 16; msk <<= 1)
          mx = fmaxf(mx, __shfl_xor(mx, msk, 32));
        const float mnew = fmaxf(Mr[qt][r], mx);
        alpha[r] = __expf(Mr[qt][r] - mnew);
        Mr[qt][r] = mnew;
        const float p0 = __expf(s0[r] - mnew);
        const float p1 = __expf(s1[r] - mnew);
        s0[r] = p0;
        s1[r] = p1;
        float rs = p0 + p1;
#pragma unroll
        for (int msk = 1; msk < 16; msk <<= 1) rs += __shfl_xor(rs, msk, 32);
        Lr[qt][r] = Lr[qt][r] * alpha[r] + rs;
#pragma unroll
        for (int nt = 0; nt < 4; ++nt) oacc[qt][nt][r] *= alpha[r];
      }

      // ---- P: C/D layout -> LDS -> A layout (per-wave buffer reused) ----
#pragma unroll
      for (int r = 0; r < 8; ++r) {
        pl[(r + 8 * h) * 32 + ml] = (bf16)s0[r];
        pl[(r + 8 * h) * 32 + 16 + ml] = (bf16)s1[r];
      }
      asm volatile("s_wait_dscnt 0" ::: "memory");  // wave-internal LDS RAW
      v16bf pa;
#pragma unroll
      for (int e = 0; e < 16; ++e) pa[e] = pl[ml * 32 + a_k(e, h)];

      // ---- O += P * V : V B operand from LDS (lane = d, elem = key) ----
#pragma unroll
      for (int nt = 0; nt < 4; ++nt) {
        const bf16* vp = vs + nt * 16 + ml;
        v16bf bv;
#pragma unroll
        for (int e = 0; e < 16; ++e) bv[e] = vp[a_k(e, h) * 64];
        oacc[qt][nt] = WMMA_BF16(pa, bv, oacc[qt][nt]);
      }
    }
    __syncthreads();  // all reads of buf[cur] done before it is re-staged
  }

  // ---- normalize and store to O laid out as (B,T,E) bf16 ----
  const int b = bh / Hh, hh = bh % Hh;
#pragma unroll
  for (int qt = 0; qt < 2; ++qt) {
#pragma unroll
    for (int nt = 0; nt < 4; ++nt) {
      const int d = nt * 16 + ml;
#pragma unroll
      for (int r = 0; r < 8; ++r) {
        const int q = q0 + qt * 16 + r + 8 * h;
        const float val = oacc[qt][nt][r] / Lr[qt][r];
        O[((size_t)b * Tt + q) * Ee + hh * Dd + d] = (bf16)val;
      }
    }
  }
}

// ---------------------------------------------------------------------------
// Kernel 3: out = attn_bf16 @ w_out_bf16 + b_out  (fp32 output).
// One wave computes 16(M) x 128(N).
// ---------------------------------------------------------------------------
__global__ __launch_bounds__(256) void out_proj(
    const bf16* __restrict__ A, const bf16* __restrict__ Wb,
    const float* __restrict__ bias, float* __restrict__ out) {
  const int lane = threadIdx.x & 31;
  const int wave = threadIdx.x >> 5;
  const int W = blockIdx.x * 8 + wave;
  const int NT = Ee / 128;  // 8 n-groups
  const int mt = W / NT, ng = W % NT;
  const int m0 = mt * 16, n0 = ng * 128;
  const int h = lane >> 4, ml = lane & 15;
  const int row = m0 + ml;

  v8f acc[8];
#pragma unroll
  for (int t = 0; t < 8; ++t) acc[t] = zero8();

  for (int kc = 0; kc < Ee / 32; ++kc) {
    v16bf a;
#pragma unroll
    for (int e = 0; e < 16; ++e)
      a[e] = A[(size_t)row * Ee + kc * 32 + a_k(e, h)];

    const bf16* bp = Wb + (size_t)(kc * 32) * Ee + n0 + ml;
#pragma unroll
    for (int t = 0; t < 8; ++t) {
      v16bf b;
#pragma unroll
      for (int e = 0; e < 16; ++e) b[e] = bp[(size_t)a_k(e, h) * Ee + t * 16];
      acc[t] = WMMA_BF16(a, b, acc[t]);
    }
  }

#pragma unroll
  for (int t = 0; t < 8; ++t) {
    const int n = n0 + t * 16 + ml;
    const float bv = bias[n];
#pragma unroll
    for (int r = 0; r < 8; ++r) {
      const int gm = m0 + r + 8 * h;
      out[(size_t)gm * Ee + n] = acc[t][r] + bv;
    }
  }
}

// ---------------------------------------------------------------------------
// Host-side launcher (graph-capture safe: only kernel launches on stream).
// ---------------------------------------------------------------------------
extern "C" void kernel_launch(void* const* d_in, const int* in_sizes, int n_in,
                              void* d_out, int out_size, void* d_ws,
                              size_t ws_size, hipStream_t stream) {
  (void)in_sizes;
  (void)n_in;
  (void)out_size;
  (void)ws_size;
  const float* x = (const float*)d_in[0];
  const float* w_qkv = (const float*)d_in[1];
  const float* b_qkv = (const float*)d_in[2];
  const float* w_out = (const float*)d_in[3];
  const float* b_out = (const float*)d_in[4];
  float* out = (float*)d_out;

  // Workspace layout (bytes):
  //   wqkv_bf 6,291,456 | wout_bf 2,097,152 | Q,K,V,O bf16 16 MB each
  char* ws = (char*)d_ws;
  bf16* wqb = (bf16*)ws;
  bf16* wob = (bf16*)(ws + 6291456);
  bf16* Qb = (bf16*)(ws + 6291456 + 2097152);
  const size_t hds = (size_t)Bb * Hh * Tt * Dd;  // 8,388,608 elems
  bf16* Kb = Qb + hds;
  bf16* Vb = Kb + hds;
  bf16* Ob = Vb + hds;

  cvt_weights<<<1024, 256, 0, stream>>>(w_qkv, w_out, wqb, wob);
  qkv_gemm<<<(Mrows / 16) * (N3 / 128) / 8, 256, 0, stream>>>(x, wqb, b_qkv,
                                                              Qb, Kb, Vb);
  flash_attn<<<dim3(Tt / 32 / 4, Bb * Hh), 128, 0, stream>>>(Qb, Kb, Vb, Ob);
  out_proj<<<(Mrows / 16) * (Ee / 128) / 8, 256, 0, stream>>>(Ob, wob, b_out,
                                                              out);
}
